// CLIPModelShuffleAttentionPenultimateLayer_59837484368490
// MI455X (gfx1250) — matrix operand order repaired and checked by reading, not compiled, not run on hardware
//
#include <hip/hip_runtime.h>
#include <hip/hip_bf16.h>

typedef __bf16 bf16_t;
typedef bf16_t v16bf  __attribute__((ext_vector_type(16)));
typedef bf16_t bf16x8 __attribute__((ext_vector_type(8)));
typedef float  v8f    __attribute__((ext_vector_type(8)));

#define BB   64
#define CC   3
#define HH   224
#define DD   1024
#define K1   150528   // C*H*W
#define PSZ  14
#define KSPL 8        // K-split for the big GEMM

__device__ __forceinline__ bf16_t f2bf(float f) { return (bf16_t)f; }

// ---------------------------------------------------------------------------
// Kernel 1: patch shuffle (deterministic hash stand-in for threefry) + fp32->bf16
// ---------------------------------------------------------------------------
__global__ void shuffle_to_bf16(const float* __restrict__ x, bf16_t* __restrict__ A) {
    size_t idx = (size_t)blockIdx.x * blockDim.x + threadIdx.x;
    const size_t total = (size_t)BB * CC * HH * HH;
    if (idx >= total) return;
    int w = (int)(idx % HH); size_t t = idx / HH;
    int h = (int)(t % HH);   t /= HH;
    int c = (int)(t % CC);   int b = (int)(t / CC);

    int ph = h / PSZ, pw = w / PSZ, r = h % PSZ, cl = w % PSZ;
    int i   = ph * 16 + pw;
    int src = (i * 101 + 37) & 255;

    unsigned s = (unsigned)(b * 256 + i) * 2654435761u + 42u;
    s ^= s >> 16; s *= 2246822519u; s ^= s >> 13;
    int kk = s & 3, f1 = (s >> 2) & 1, f2 = (s >> 3) & 1;

    int r1 = f1 ? (PSZ - 1 - r)  : r;
    int c1 = f2 ? (PSZ - 1 - cl) : cl;
    int rs, cs;
    switch (kk) {
        case 0:  rs = r1;            cs = c1;            break;
        case 1:  rs = c1;            cs = PSZ - 1 - r1;  break;
        case 2:  rs = PSZ - 1 - r1;  cs = PSZ - 1 - c1;  break;
        default: rs = PSZ - 1 - c1;  cs = r1;            break;
    }
    int hs = (src / 16) * PSZ + rs;
    int ws = (src % 16) * PSZ + cs;
    float val = x[(((size_t)b * CC + c) * HH + hs) * HH + ws];
    A[idx] = f2bf(val);
}

// ---------------------------------------------------------------------------
// Kernel 2: WMMA GEMM, N compile-time so B loads use immediate offsets.
//   C(f32) = A(bf16, MxK) * B(f32->bf16 in-flight, KxN)
// One wave per 16x16 tile (x KSPLIT partials). Pointers strength-reduced.
// ---------------------------------------------------------------------------
template <int N, int NSPLIT>
__global__ void wmma_gemm(const bf16_t* __restrict__ A,
                          const float*  __restrict__ Bm,
                          float*        __restrict__ C,
                          int M, int K) {
    const int lane  = threadIdx.x & 31;
    const int wave  = threadIdx.x >> 5;
    const int tile  = blockIdx.x * (blockDim.x >> 5) + wave;
    const int mtls  = M >> 4;
    const int ntls  = N >> 4;
    if (tile >= mtls * ntls * NSPLIT) return;
    const int sp = tile / (mtls * ntls);
    const int t2 = tile % (mtls * ntls);
    const int mt = t2 / ntls;
    const int nt = t2 % ntls;

    const int row  = lane & 15;          // A row in tile / B col (n) in tile
    const int hi   = (lane >> 4) << 3;   // 0 or 8
    const int kgrp = (lane >> 4) << 4;   // 0 or 16

    const int klen = K / NSPLIT;         // multiple of 32
    const int kbeg = sp * klen;

    const bf16_t* ap = A  + (size_t)(mt * 16 + row) * K + kbeg + kgrp;
    const float*  bp = Bm + (size_t)(kbeg + kgrp) * N + nt * 16 + row;

    v8f acc = {};
    for (int k = 0; k < klen; k += 32) {
        bf16x8 a0 = *(const bf16x8*)(ap);
        bf16x8 a1 = *(const bf16x8*)(ap + 8);
        v16bf a;
#pragma unroll
        for (int e = 0; e < 8; ++e) { a[e] = a0[e]; a[8 + e] = a1[e]; }

        float bfv[16];
#pragma unroll
        for (int e = 0; e < 16; ++e) bfv[e] = bp[e * N];   // immediate offsets
        __builtin_prefetch((const void*)(bp + 32 * N), 0, 0);
        v16bf b;
#pragma unroll
        for (int e = 0; e < 16; ++e) b[e] = f2bf(bfv[e]);

        acc = __builtin_amdgcn_wmma_f32_16x16x32_bf16(
                  false, a, false, b, (short)0, acc, false, false);
        ap += 32;
        bp += (size_t)32 * N;
    }

    float* Cp = C + (size_t)sp * M * N + (size_t)(mt * 16 + hi) * N + nt * 16 + row;
#pragma unroll
    for (int r = 0; r < 8; ++r) Cp[r * N] = acc[r];
}

// Deterministic sum of KSPL partial C matrices.
__global__ void reduce_partials(const float* __restrict__ part,
                                float* __restrict__ outb, int n) {
    int idx = blockIdx.x * blockDim.x + threadIdx.x;
    if (idx >= n) return;
    float s = 0.f;
#pragma unroll
    for (int p = 0; p < KSPL; ++p) s += part[(size_t)p * n + idx];
    outb[idx] = s;
}

// ---------------------------------------------------------------------------
__global__ void assemble_stack(const float* __restrict__ fs,
                               const float* __restrict__ fo,
                               const float* __restrict__ nf,
                               bf16_t* __restrict__ st) {
    int idx = blockIdx.x * blockDim.x + threadIdx.x;
    if (idx >= BB * DD) return;
    int b = idx >> 10, d = idx & (DD - 1);
    st[((size_t)b * 3 + 0) * DD + d] = f2bf(fs[idx]);
    st[((size_t)b * 3 + 1) * DD + d] = f2bf(fo[idx]);
    st[((size_t)b * 3 + 2) * DD + d] = f2bf(nf[idx]);
}

__device__ __forceinline__ float block_reduce_sum(float v, float* red, int t, int n) {
    red[t] = v; __syncthreads();
    for (int off = n >> 1; off > 0; off >>= 1) {
        if (t < off) red[t] += red[t + off];
        __syncthreads();
    }
    float r = red[0]; __syncthreads();
    return r;
}

// ---------------------------------------------------------------------------
__global__ void attention_avg(const float* __restrict__ q,
                              const float* __restrict__ k,
                              const float* __restrict__ v,
                              bf16_t* __restrict__ avg_bf) {
    const int b = blockIdx.x, t = threadIdx.x;
    __shared__ float red[256];
    __shared__ float S[9];
    __shared__ float wgt[3];
    const float* qb = q + (size_t)b * 3 * DD;
    const float* kb = k + (size_t)b * 3 * DD;
    for (int p = 0; p < 9; ++p) {
        int n = p / 3, m = p % 3;
        float s = 0.f;
        for (int d = t; d < DD; d += 256) s += qb[n * DD + d] * kb[m * DD + d];
        float tot = block_reduce_sum(s, red, t, 256);
        if (t == 0) S[p] = tot * 0.03125f;   // 1/sqrt(1024)
    }
    __syncthreads();
    if (t == 0) {
        float w0 = 0.f, w1 = 0.f, w2 = 0.f;
        for (int n = 0; n < 3; ++n) {
            float s0 = S[n * 3], s1 = S[n * 3 + 1], s2 = S[n * 3 + 2];
            float mx = fmaxf(s0, fmaxf(s1, s2));
            float e0 = __expf(s0 - mx), e1 = __expf(s1 - mx), e2 = __expf(s2 - mx);
            float inv = 1.f / (e0 + e1 + e2);
            w0 += e0 * inv; w1 += e1 * inv; w2 += e2 * inv;
        }
        const float third = 1.f / 3.f;
        wgt[0] = w0 * third; wgt[1] = w1 * third; wgt[2] = w2 * third;
    }
    __syncthreads();
    const float* vb = v + (size_t)b * 3 * DD;
    for (int d = t; d < DD; d += 256) {
        float s = wgt[0] * vb[d] + wgt[1] * vb[DD + d] + wgt[2] * vb[2 * DD + d];
        avg_bf[(size_t)b * DD + d] = f2bf(s);
    }
}

// ---------------------------------------------------------------------------
__global__ void moments_kernel(const float* __restrict__ reduced,
                               float* __restrict__ moments) {
    const int b = blockIdx.x, j = threadIdx.x;       // j < 256
    const float* rb = reduced + (size_t)b * 3 * 256 + j;
    float x0 = rb[0], x1 = rb[256], x2 = rb[512];
    const float s13 = 0.5773502691896258f;
    const float s23 = 0.8164965809277260f;
    const float ca  = 0.8660254037844387f;
    float d0 = s13 * (x0 + x1 + x2);
    float d1 = s23 * (ca * x0 - ca * x2);
    float d2 = s23 * (0.5f * x0 - x1 + 0.5f * x2);
    const float third = 1.f / 3.f;
    float mean = (d0 + d1 + d2) * third;
    float a0 = d0 - mean, a1 = d1 - mean, a2 = d2 - mean;
    float var  = (a0 * a0 + a1 * a1 + a2 * a2) * third;
    float sd   = sqrtf(var + 1e-5f);
    float m3   = (a0 * a0 * a0 + a1 * a1 * a1 + a2 * a2 * a2) * third;
    float m4   = (a0 * a0 * a0 * a0 + a1 * a1 * a1 * a1 + a2 * a2 * a2 * a2) * third;
    float skew = m3 / (sd * sd * sd + 1e-5f);
    float kurt = m4 / (sd * sd * sd * sd + 1e-5f);

    __shared__ float red[256];
    float ss = mean * mean + var * var + skew * skew + kurt * kurt;
    float tot = block_reduce_sum(ss, red, j, 256);
    float inv = 1.f / fmaxf(sqrtf(tot), 1e-12f);
    float* mo = moments + (size_t)b * DD;
    mo[j]        = mean * inv;
    mo[256 + j]  = var  * inv;
    mo[512 + j]  = skew * inv;
    mo[768 + j]  = kurt * inv;
}

// ---------------------------------------------------------------------------
__global__ void finalize_kernel(const float* __restrict__ attn_raw,
                                const float* __restrict__ moments,
                                const float* __restrict__ fc_w,
                                const float* __restrict__ fc_b,
                                float* __restrict__ out) {
    const int b = blockIdx.x, t = threadIdx.x;
    __shared__ float red[256];
    const float* ar = attn_raw + (size_t)b * DD;
    float ss = 0.f;
    for (int d = t; d < DD; d += 256) { float x = ar[d]; ss += x * x; }
    float nrm = block_reduce_sum(ss, red, t, 256);
    float inv = 1.f / fmaxf(sqrtf(nrm), 1e-12f);
    const float* mo = moments + (size_t)b * DD;
    float dot = 0.f;
    for (int d = t; d < DD; d += 256) dot += ar[d] * inv * fc_w[d];
    for (int d = t; d < DD; d += 256) dot += mo[d] * fc_w[DD + d];
    float tot = block_reduce_sum(dot, red, t, 256);
    if (t == 0) out[b] = tot + fc_b[0];
}

// ---------------------------------------------------------------------------
extern "C" void kernel_launch(void* const* d_in, const int* in_sizes, int n_in,
                              void* d_out, int out_size, void* d_ws, size_t ws_size,
                              hipStream_t stream) {
    const float* x         = (const float*)d_in[0];
    const float* feat_orig = (const float*)d_in[1];
    const float* noise     = (const float*)d_in[2];
    const float* W_clip    = (const float*)d_in[3];
    const float* W_red     = (const float*)d_in[4];
    const float* Wq        = (const float*)d_in[5];
    const float* Wk        = (const float*)d_in[6];
    const float* Wv        = (const float*)d_in[7];
    const float* Wo        = (const float*)d_in[8];
    const float* fc_w      = (const float*)d_in[9];
    const float* fc_b      = (const float*)d_in[10];
    float* out             = (float*)d_out;

    char* ws = (char*)d_ws;
    size_t off = 0;
    auto alloc = [&](size_t bytes) { size_t o = off; off = (off + bytes + 255) & ~(size_t)255; return o; };
    bf16_t* A_shuf   = (bf16_t*)(ws + alloc((size_t)BB * K1 * 2));            // 19.3 MB
    float*  part     = (float*) (ws + alloc((size_t)KSPL * BB * DD * 4));     // 2 MB
    float*  feat_sh  = (float*) (ws + alloc((size_t)BB * DD * 4));
    bf16_t* stack_bf = (bf16_t*)(ws + alloc((size_t)BB * 3 * DD * 2));
    float*  qbuf     = (float*) (ws + alloc((size_t)BB * 3 * DD * 4));
    float*  kbuf     = (float*) (ws + alloc((size_t)BB * 3 * DD * 4));
    float*  vbuf     = (float*) (ws + alloc((size_t)BB * 3 * DD * 4));
    float*  reduced  = (float*) (ws + alloc((size_t)BB * 3 * 256 * 4));
    bf16_t* avg_bf   = (bf16_t*)(ws + alloc((size_t)BB * DD * 2));
    float*  attn_raw = (float*) (ws + alloc((size_t)BB * DD * 4));
    float*  moments  = (float*) (ws + alloc((size_t)BB * DD * 4));

    // 1) shuffled patches -> bf16 A matrix
    {
        size_t total = (size_t)BB * CC * HH * HH;
        shuffle_to_bf16<<<(int)((total + 255) / 256), 256, 0, stream>>>(x, A_shuf);
    }
    // 2) feat_shuffle = A_shuf @ W_clip, K-split x8 for HBM latency hiding
    {
        int tiles = (BB / 16) * (DD / 16) * KSPL;              // 2048 waves
        wmma_gemm<DD, KSPL><<<(tiles + 3) / 4, 128, 0, stream>>>(A_shuf, W_clip, part, BB, K1);
        reduce_partials<<<(BB * DD + 255) / 256, 256, 0, stream>>>(part, feat_sh, BB * DD);
    }
    // 3) stack bf16
    assemble_stack<<<(BB * DD + 255) / 256, 256, 0, stream>>>(feat_sh, feat_orig, noise, stack_bf);
    // 4) q,k,v and reduced GEMMs (192 x 1024 x {1024,256})
    {
        int M = BB * 3;
        int tqkv = (M / 16) * (DD / 16);                       // 768
        wmma_gemm<DD, 1><<<(tqkv + 3) / 4, 128, 0, stream>>>(stack_bf, Wq, qbuf, M, DD);
        wmma_gemm<DD, 1><<<(tqkv + 3) / 4, 128, 0, stream>>>(stack_bf, Wk, kbuf, M, DD);
        wmma_gemm<DD, 1><<<(tqkv + 3) / 4, 128, 0, stream>>>(stack_bf, Wv, vbuf, M, DD);
        int tred = (M / 16) * (256 / 16);                      // 192
        wmma_gemm<256, 1><<<(tred + 3) / 4, 128, 0, stream>>>(stack_bf, W_red, reduced, M, DD);
    }
    // 5) attention (softmax commuted with mean over n)
    attention_avg<<<BB, 256, 0, stream>>>(qbuf, kbuf, vbuf, avg_bf);
    // 6) attn_out = avg_v @ Wo
    {
        int tiles = (BB / 16) * (DD / 16);
        wmma_gemm<DD, 1><<<(tiles + 3) / 4, 128, 0, stream>>>(avg_bf, Wo, attn_raw, BB, DD);
    }
    // 7) DCT moments + l2norm
    moments_kernel<<<BB, 256, 0, stream>>>(reduced, moments);
    // 8) l2norm(attn_out) + fc
    finalize_kernel<<<BB, 256, 0, stream>>>(attn_raw, moments, fc_w, fc_b, out);
}
